// AttentionGate_16020228014534
// MI455X (gfx1250) — matrix operand order repaired and checked by reading, not compiled, not run on hardware
//
#include <hip/hip_runtime.h>
#include <hip/hip_bf16.h>

// AttentionGate: B=2, CF=CG=64, CI=32, D=H=W=64 (SP = 2^18 voxels per batch)
//   g1 = Wg@g, x1 = Wx@x  (per-voxel 32x64 matvec)  -> WMMA f16, f32 accum
//   a  = relu(g1+x1); psi = sigmoid(Wpsi@a); out = x*psi
//
// Strategy (memory-bound, ~400MB HBM traffic -> ~17us floor @23.3TB/s):
//  - pack [Wg|Wx] into per-lane v_wmma A-fragment layout once (tiny kernel)
//  - each wave32 handles a 16-voxel tile: 8x v_wmma_f32_16x16x32_f16 computes
//    the fused 32x16 pre-activation (K = 64 g-channels + 64 x-channels)
//  - psi via C-layout-aware dot + __shfl_xor(16) cross-half reduce + sigmoid
//  - final x*psi pass uses float4 loads/stores (coalesced 64B segments)

typedef __attribute__((ext_vector_type(16))) _Float16 v16h;
typedef __attribute__((ext_vector_type(8)))  float    v8f;

#define SP_SHIFT 18
#define SP (1 << SP_SHIFT)          // 64*64*64
#define NB 2
#define NVOX (NB * SP)              // 524288 voxels
#define WAVES_PER_BLOCK 8

union V16H { v16h v; _Float16 h[16]; };

// ---------------------------------------------------------------------------
// Pack Wg/Wx (f32 [32][64]) into f16 WMMA A-fragments, one v16h per lane.
// Fragment f: mt = f&1 (M-tile 0/1), inp = (f>>1)&1 (0=Wg,1=Wx), kb = f>>2.
// A-fragment element e of lane L (h=L>>4): K = h*16 + e
// (contiguous k order, matching the B fragment fill).
// ---------------------------------------------------------------------------
__global__ void pack_weights_kernel(const float* __restrict__ Wg,
                                    const float* __restrict__ Wx,
                                    _Float16* __restrict__ wpack) {
  const int t    = threadIdx.x;      // 0..255 = 8 fragments x 32 lanes
  const int f    = t >> 5;
  const int lane = t & 31;
  const int mt   = f & 1;
  const float* W = ((f >> 1) & 1) ? Wx : Wg;
  const int kb   = (f >> 2) * 32;
  const int row  = (lane & 15) + 16 * mt;   // ci row 0..31
  const int h    = lane >> 4;
  _Float16* dst  = wpack + (f * 32 + lane) * 16;
#pragma unroll
  for (int e = 0; e < 16; ++e) {
    const int K = kb + h * 16 + e;   // channel 0..63
    dst[e] = (_Float16)W[row * 64 + K];
  }
}

// ---------------------------------------------------------------------------
// Main fused kernel: one wave per 16-voxel tile, 8 waves per block.
// ---------------------------------------------------------------------------
__global__ void attention_gate_kernel(const float* __restrict__ x,
                                      const float* __restrict__ g,
                                      const float* __restrict__ Wpsi,
                                      const _Float16* __restrict__ wpack,
                                      float* __restrict__ out) {
  const int lane = threadIdx.x & 31;
  const int wv   = threadIdx.x >> 5;
  const int tile = blockIdx.x * WAVES_PER_BLOCK + wv;
  const long v0  = (long)tile * 16;             // tile never crosses batch (SP%16==0)
  const int  b   = (int)(v0 >> SP_SHIFT);
  const int  sp0 = (int)(v0 & (SP - 1));
  const int  col = lane & 15;                   // B/C column = voxel within tile
  const int  hh  = lane >> 4;                   // lane half

  // --- A fragments: one coalesced 32B read per fragment ---
  const v16h* wp = (const v16h*)wpack;
  v16h A[8];
#pragma unroll
  for (int f = 0; f < 8; ++f) A[f] = wp[f * 32 + lane];

  // --- B fragments: contiguous k order (as in A pack): lane holds column `col`,
  //     element e holds K = kbase + hh*16 + e (channel), f32->f16 on load ---
  auto loadB = [&](const float* __restrict__ src, int kbase) -> v16h {
    V16H r;
    const float* p = src + (((long)(b * 64 + kbase + hh * 16)) << SP_SHIFT) + sp0 + col;
#pragma unroll
    for (int e = 0; e < 16; ++e) r.h[e] = (_Float16)p[(long)e << SP_SHIFT];
    return r.v;
  };
  const v16h Bg0 = loadB(g, 0);
  const v16h Bg1 = loadB(g, 32);
  const v16h Bx0 = loadB(x, 0);
  const v16h Bx1 = loadB(x, 32);

  // --- fused pre-activation: acc[32x16] = Wg@g + Wx@x over K=128 total ---
  v8f acc0 = {};   // rows 0..15 of CI
  v8f acc1 = {};   // rows 16..31 of CI
  acc0 = __builtin_amdgcn_wmma_f32_16x16x32_f16(false, A[0], false, Bg0, (short)0, acc0, false, false);
  acc1 = __builtin_amdgcn_wmma_f32_16x16x32_f16(false, A[1], false, Bg0, (short)0, acc1, false, false);
  acc0 = __builtin_amdgcn_wmma_f32_16x16x32_f16(false, A[4], false, Bg1, (short)0, acc0, false, false);
  acc1 = __builtin_amdgcn_wmma_f32_16x16x32_f16(false, A[5], false, Bg1, (short)0, acc1, false, false);
  acc0 = __builtin_amdgcn_wmma_f32_16x16x32_f16(false, A[2], false, Bx0, (short)0, acc0, false, false);
  acc1 = __builtin_amdgcn_wmma_f32_16x16x32_f16(false, A[3], false, Bx0, (short)0, acc1, false, false);
  acc0 = __builtin_amdgcn_wmma_f32_16x16x32_f16(false, A[6], false, Bx1, (short)0, acc0, false, false);
  acc1 = __builtin_amdgcn_wmma_f32_16x16x32_f16(false, A[7], false, Bx1, (short)0, acc1, false, false);

  // --- psi = sigmoid( Wpsi . relu(acc) ) ---
  // C layout: VGPR r of lane L holds row (hh*8 + r) [acc0] / 16+hh*8+r [acc1],
  // column = col. Each lane sums its 16 rows; xor-16 shuffle adds the other half.
  float partial = 0.0f;
#pragma unroll
  for (int r = 0; r < 8; ++r) {
    const float wlo = Wpsi[hh * 8 + r];
    const float whi = Wpsi[16 + hh * 8 + r];
    float a0 = acc0[r]; a0 = a0 > 0.0f ? a0 : 0.0f;
    float a1 = acc1[r]; a1 = a1 > 0.0f ? a1 : 0.0f;
    partial = fmaf(wlo, a0, partial);
    partial = fmaf(whi, a1, partial);
  }
  partial += __shfl_xor(partial, 16, 32);
  const float psi = 1.0f / (1.0f + __expf(-partial));   // psi for voxel sp0+col

  // --- out[c, v] = x[c, v] * psi[v], float4-vectorized ---
  // lane (q = lane&3) covers voxels sp0 + 4q .. +3; gather those 4 psi values.
  const int q = lane & 3;
  float4 psi4;
  psi4.x = __shfl(psi, q * 4 + 0, 32);
  psi4.y = __shfl(psi, q * 4 + 1, 32);
  psi4.z = __shfl(psi, q * 4 + 2, 32);
  psi4.w = __shfl(psi, q * 4 + 3, 32);

  const int crow = lane >> 2;                  // 8 channels per iteration
#pragma unroll
  for (int it = 0; it < 8; ++it) {
    const int c = it * 8 + crow;
    const long idx = (((long)(b * 64 + c)) << SP_SHIFT) + sp0 + q * 4;
    const float4 xv = *(const float4*)(x + idx);
    float4 o;
    o.x = xv.x * psi4.x;
    o.y = xv.y * psi4.y;
    o.z = xv.z * psi4.z;
    o.w = xv.w * psi4.w;
    *(float4*)(out + idx) = o;
  }
}

extern "C" void kernel_launch(void* const* d_in, const int* in_sizes, int n_in,
                              void* d_out, int out_size, void* d_ws, size_t ws_size,
                              hipStream_t stream) {
  const float* x    = (const float*)d_in[0];   // [2,64,64,64,64]
  const float* g    = (const float*)d_in[1];   // [2,64,64,64,64]
  const float* Wg   = (const float*)d_in[2];   // [32,64]
  const float* Wx   = (const float*)d_in[3];   // [32,64]
  const float* Wpsi = (const float*)d_in[4];   // [1,32]
  float* out        = (float*)d_out;
  _Float16* wpack   = (_Float16*)d_ws;         // 8 frags * 32 lanes * 16 f16 = 8KB

  pack_weights_kernel<<<1, 256, 0, stream>>>(Wg, Wx, wpack);

  const int tiles = NVOX / 16;                 // 32768 wave-tiles
  attention_gate_kernel<<<tiles / WAVES_PER_BLOCK, WAVES_PER_BLOCK * 32, 0, stream>>>(
      x, g, Wpsi, wpack, out);
}
